// EdgeProbMLP_89781996355945
// MI455X (gfx1250) — matrix operand order repaired and checked by reading, not compiled, not run on hardware
//
#include <hip/hip_runtime.h>
#include <hip/hip_bf16.h>
#include <cstdint>

typedef __attribute__((ext_vector_type(16))) _Float16 v16h;
typedef __attribute__((ext_vector_type(8)))  _Float16 v8h;
typedef __attribute__((ext_vector_type(8)))  float    v8f;
typedef __attribute__((ext_vector_type(4)))  float    v4f;

#define IN_CH 128
#define HID   128

static __device__ __forceinline__ v8f wmma_f16(v16h a, v16h b, v8f c) {
    return __builtin_amdgcn_wmma_f32_16x16x32_f16(
        /*neg_a=*/false, a, /*neg_b=*/false, b,
        /*c_mod=*/(short)0, c, /*reuse_a=*/false, /*reuse_b=*/false);
}

// ---------------------------------------------------------------------------
// Kernel 1:  P[node][128] = f16( relu( nodes @ Wdim^T + bdim ) )
// One wave computes a 16x128 output tile via 4 (K) x 8 (N) WMMA f16 ops.
// Wdim pre-swizzled into LDS as the B operand (contiguous K, same order as A):
//   B[k][n] = Wdim[n][k];  tile (kt, nt): lane<16 holds col n=nt*16+lane,
//   K = kt*32 + 0..15 contiguous; lane>=16 holds K = kt*32 + 16..31.
// ---------------------------------------------------------------------------
__global__ __launch_bounds__(256)
void node_proj_kernel(const float* __restrict__ nodes,
                      const float* __restrict__ Wdim,
                      const float* __restrict__ bdim,
                      _Float16*    __restrict__ P,
                      int n_nodes)
{
    __shared__ __align__(32) _Float16 Bl[4 * 8 * 32 * 16];   // 32 KB

    const int tid = threadIdx.x;
    for (int i = tid; i < 4 * 8 * 32; i += 256) {
        const int lane = i & 31, nt = (i >> 5) & 7, kt = i >> 8;
        const int n  = nt * 16 + (lane & 15);
        const int kb = kt * 32 + ((lane & 16) ? 16 : 0);
        const float* src = Wdim + n * IN_CH + kb;          // contiguous 16 f32
        #pragma unroll
        for (int j = 0; j < 16; ++j) Bl[i * 16 + j] = (_Float16)src[j];
    }
    __syncthreads();

    const int wave = tid >> 5, lane = tid & 31;
    const int m    = lane & 15;        // row of A held by this lane
    const int hi   = lane >> 4;        // lane-half selects K striping
    const int row0 = blockIdx.x * 128 + wave * 16;

    int r = row0 + m; if (r > n_nodes - 1) r = n_nodes - 1;   // clamp, keep EXEC full
    const float* arow = nodes + (size_t)r * IN_CH;

    v8f zero = {};
    v8f acc[8];
    #pragma unroll
    for (int nt = 0; nt < 8; ++nt) acc[nt] = zero;

    #pragma unroll
    for (int kt = 0; kt < 4; ++kt) {
        // A order used here (contiguous K, matches B): lane<16 -> K kt*32+{0..15};
        //                          lane>=16 -> K kt*32+{16..31}
        const int o1 = kt * 32 + hi * 16;
        const int o2 = kt * 32 + 8 + hi * 16;
        v4f f0 = *(const v4f*)(arow + o1);
        v4f f1 = *(const v4f*)(arow + o1 + 4);
        v4f f2 = *(const v4f*)(arow + o2);
        v4f f3 = *(const v4f*)(arow + o2 + 4);
        v16h a;
        #pragma unroll
        for (int j = 0; j < 4; ++j) {
            a[j]      = (_Float16)f0[j];
            a[4 + j]  = (_Float16)f1[j];
            a[8 + j]  = (_Float16)f2[j];
            a[12 + j] = (_Float16)f3[j];
        }
        #pragma unroll
        for (int nt = 0; nt < 8; ++nt) {
            const v16h b = *(const v16h*)(Bl + ((kt * 8 + nt) * 32 + lane) * 16);
            acc[nt] = wmma_f16(a, b, acc[nt]);
        }
    }

    // C/D layout: lane = column N (nt*16 + m), VGPR idx rr + 8*hi = row M.
    #pragma unroll
    for (int nt = 0; nt < 8; ++nt) {
        const int   n    = nt * 16 + m;
        const float bias = bdim[n];
        #pragma unroll
        for (int rr = 0; rr < 8; ++rr) {
            const int row = row0 + rr + hi * 8;
            if (row < n_nodes) {
                float v = acc[nt][rr] + bias;
                v = v > 0.0f ? v : 0.0f;
                P[(size_t)row * HID + n] = (_Float16)v;
            }
        }
    }
}

// ---------------------------------------------------------------------------
// Kernel 2: one wave processes TWO 16-edge tiles (M = 32) so every B tile
// fetched from LDS feeds two back-to-back independent WMMAs (halves
// ds traffic + s_wait_dscnt stalls per matrix op).
//   x = P[src], y = P[dst]   (f16, L2-resident gathers)
//   A[32x256] = [x*y | x-y]  (packed f16 VALU, built in WMMA A layout)
//   H[32x128] = A @ W1^T     (128 WMMA f16 ops, W1 swizzled in LDS)
//   out[e]    = sigmoid( relu(H + b1) . W2 + b2 )  (shfl-xor reduction)
// ---------------------------------------------------------------------------
__global__ __launch_bounds__(256)
void edge_mlp_kernel(const int*      __restrict__ eidx,
                     const _Float16* __restrict__ P,
                     const float*    __restrict__ W1,
                     const float*    __restrict__ b1,
                     const float*    __restrict__ W2,
                     const float*    __restrict__ b2,
                     float*          __restrict__ out,
                     int n_edges)
{
    __shared__ __align__(32) _Float16 Bl[8 * 8 * 32 * 16];   // 64 KB: W1 B-layout

    const int tid = threadIdx.x;
    for (int i = tid; i < 8 * 8 * 32; i += 256) {
        const int lane = i & 31, nt = (i >> 5) & 7, kt = i >> 8;
        const int n  = nt * 16 + (lane & 15);
        const int kb = kt * 32 + ((lane & 16) ? 16 : 0);
        const float* src = W1 + n * (2 * HID) + kb;        // contiguous 16 f32
        #pragma unroll
        for (int j = 0; j < 16; ++j) Bl[i * 16 + j] = (_Float16)src[j];
    }
    __syncthreads();

    const int wave   = tid >> 5, lane = tid & 31;
    const int ntiles = (n_edges + 15) >> 4;
    const int tile0  = (blockIdx.x * 8 + wave) * 2;
    if (tile0 >= ntiles) return;                 // whole-wave exit: EXEC stays full
    const int tile1 = tile0 + 1;

    const int m  = lane & 15;
    const int hi = lane >> 4;
    int e0 = tile0 * 16 + m; if (e0 > n_edges - 1) e0 = n_edges - 1;
    int e1 = tile1 * 16 + m; if (e1 > n_edges - 1) e1 = n_edges - 1;
    const int s0 = eidx[e0], d0 = eidx[n_edges + e0];
    const int s1 = eidx[e1], d1 = eidx[n_edges + e1];
    const _Float16* xr0 = P + (size_t)s0 * HID;
    const _Float16* yr0 = P + (size_t)d0 * HID;
    const _Float16* xr1 = P + (size_t)s1 * HID;
    const _Float16* yr1 = P + (size_t)d1 * HID;

    v8f zero = {};
    v8f acc0[8], acc1[8];
    #pragma unroll
    for (int nt = 0; nt < 8; ++nt) { acc0[nt] = zero; acc1[nt] = zero; }

    #pragma unroll
    for (int kt = 0; kt < 4; ++kt) {
        const int o1 = kt * 32 + hi * 16;        // A striping within K-block
        const int o2 = kt * 32 + 8 + hi * 16;

        v8h xa0 = *(const v8h*)(xr0 + o1);
        v8h xb0 = *(const v8h*)(xr0 + o2);
        v8h ya0 = *(const v8h*)(yr0 + o1);
        v8h yb0 = *(const v8h*)(yr0 + o2);
        v8h xa1 = *(const v8h*)(xr1 + o1);
        v8h xb1 = *(const v8h*)(xr1 + o2);
        v8h ya1 = *(const v8h*)(yr1 + o1);
        v8h yb1 = *(const v8h*)(yr1 + o2);

        v16h xh0 = __builtin_shufflevector(xa0, xb0, 0,1,2,3,4,5,6,7,8,9,10,11,12,13,14,15);
        v16h yh0 = __builtin_shufflevector(ya0, yb0, 0,1,2,3,4,5,6,7,8,9,10,11,12,13,14,15);
        v16h xh1 = __builtin_shufflevector(xa1, xb1, 0,1,2,3,4,5,6,7,8,9,10,11,12,13,14,15);
        v16h yh1 = __builtin_shufflevector(ya1, yb1, 0,1,2,3,4,5,6,7,8,9,10,11,12,13,14,15);

        v16h ap0 = xh0 * yh0;   // edge_feat K-blocks 0..3  (x*y)
        v16h ad0 = xh0 - yh0;   // edge_feat K-blocks 4..7  (x-y)
        v16h ap1 = xh1 * yh1;
        v16h ad1 = xh1 - yh1;

        #pragma unroll
        for (int nt = 0; nt < 8; ++nt) {
            const v16h b = *(const v16h*)(Bl + ((kt * 8 + nt) * 32 + lane) * 16);
            acc0[nt] = wmma_f16(ap0, b, acc0[nt]);   // same B, independent D:
            acc1[nt] = wmma_f16(ap1, b, acc1[nt]);   // no hazard, b reused
        }
        #pragma unroll
        for (int nt = 0; nt < 8; ++nt) {
            const v16h b = *(const v16h*)(Bl + (((kt + 4) * 8 + nt) * 32 + lane) * 16);
            acc0[nt] = wmma_f16(ad0, b, acc0[nt]);
            acc1[nt] = wmma_f16(ad1, b, acc1[nt]);
        }
    }

    // Epilogue: bias + ReLU + dot with W2, reduce across the 16 lanes (= N dim).
    float b1v[8], w2v[8];
    #pragma unroll
    for (int nt = 0; nt < 8; ++nt) {
        const int n = nt * 16 + m;
        b1v[nt] = b1[n];
        w2v[nt] = W2[n];
    }
    const float bias2 = b2[0];

    #pragma unroll
    for (int t = 0; t < 2; ++t) {
        const v8f* acc  = (t == 0) ? acc0 : acc1;
        const int  tile = (t == 0) ? tile0 : tile1;
        #pragma unroll
        for (int rr = 0; rr < 8; ++rr) {
            float p = 0.0f;
            #pragma unroll
            for (int nt = 0; nt < 8; ++nt) {
                float h = acc[nt][rr] + b1v[nt];
                h = h > 0.0f ? h : 0.0f;
                p += h * w2v[nt];
            }
            p += __shfl_xor(p, 1, 16);
            p += __shfl_xor(p, 2, 16);
            p += __shfl_xor(p, 4, 16);
            p += __shfl_xor(p, 8, 16);
            const int eo = tile * 16 + rr + hi * 8;
            if (m == 0 && eo < n_edges) {
                const float tt = p + bias2;
                out[eo] = 1.0f / (1.0f + __expf(-tt));
            }
        }
    }
}

// ---------------------------------------------------------------------------
extern "C" void kernel_launch(void* const* d_in, const int* in_sizes, int n_in,
                              void* d_out, int out_size, void* d_ws, size_t ws_size,
                              hipStream_t stream)
{
    const float* nodes = (const float*)d_in[0];
    const int*   eidx  = (const int*)  d_in[1];
    const float* Wdim  = (const float*)d_in[2];
    const float* bdim  = (const float*)d_in[3];
    const float* W1    = (const float*)d_in[4];
    const float* b1    = (const float*)d_in[5];
    const float* W2    = (const float*)d_in[6];
    const float* b2    = (const float*)d_in[7];
    float* out = (float*)d_out;

    const int n_nodes = in_sizes[0] / IN_CH;
    const int n_edges = in_sizes[1] / 2;

    _Float16* P = (_Float16*)d_ws;    // 25.6 MB projected node table (L2-resident)

    const int blocks1 = (n_nodes + 127) / 128;
    node_proj_kernel<<<blocks1, 256, 0, stream>>>(nodes, Wdim, bdim, P, n_nodes);

    const int ntiles  = (n_edges + 15) / 16;     // 16-edge tiles
    const int npairs  = (ntiles + 1) / 2;        // 2 tiles per wave
    const int blocks2 = (npairs + 7) / 8;        // 8 waves per block
    edge_mlp_kernel<<<blocks2, 256, 0, stream>>>(eidx, P, W1, b1, W2, b2, out, n_edges);
}
